// FlowNet3D_11802570130415
// MI455X (gfx1250) — matrix operand order repaired and checked by reading, not compiled, not run on hardware
//
#include <hip/hip_runtime.h>
#include <hip/hip_bf16.h>

typedef _Float16 f16;
typedef __attribute__((ext_vector_type(16))) _Float16 v16h;
typedef __attribute__((ext_vector_type(8)))  float    v8f;

static inline int ceilDiv(int a, int b) { return (a + b - 1) / b; }
static inline int cp32(int x) { return (x + 31) & ~31; }

__device__ inline v16h mk_v16h(float4 x, float4 y) {
  union U { float4 f[2]; v16h v; } u;
  u.f[0] = x; u.f[1] = y;
  return u.v;
}

// ---------------------------------------------------------------------------
// BN folding into padded f16 weights: Weff[Opad x Cpad] (zero-padded),
// beff = (b-mean)*s + beta,  s = gamma*rsqrt(var+eps)
// ---------------------------------------------------------------------------
__global__ void k_fold(const float* __restrict__ W, const float* __restrict__ b,
                       const float* __restrict__ g, const float* __restrict__ be,
                       const float* __restrict__ mean, const float* __restrict__ var,
                       f16* __restrict__ Weff, float* __restrict__ beff,
                       int O, int C, int Cpad, int total) {
  int t = blockIdx.x * 256 + threadIdx.x;
  if (t >= total) return;
  int o = t / Cpad, c = t % Cpad;
  if (o < O && c < C) {
    float s = g[o] * rsqrtf(var[o] + 1e-3f);
    Weff[t] = (f16)(W[(size_t)o * C + c] * s);
    if (c == 0) beff[o] = (b[o] - mean[o]) * s + be[o];
  } else {
    Weff[t] = (f16)0.0f;
  }
}

__global__ void k_fold_plain(const float* __restrict__ W, const float* __restrict__ b,
                             f16* __restrict__ Weff, float* __restrict__ beff,
                             int O, int C, int Cpad, int total) {
  int t = blockIdx.x * 256 + threadIdx.x;
  if (t >= total) return;
  int o = t / Cpad, c = t % Cpad;
  if (o < O && c < C) {
    Weff[t] = (f16)W[(size_t)o * C + c];
    if (c == 0) beff[o] = b[o];
  } else {
    Weff[t] = (f16)0.0f;
  }
}

__global__ void k_f2h(const float* __restrict__ in, f16* __restrict__ out, int n) {
  int t = blockIdx.x * 256 + threadIdx.x;
  if (t < n) out[t] = (f16)in[t];
}

// ---------------------------------------------------------------------------
// WMMA GEMM:  Y[b,o,l] = act( sum_c Weff[o,c]*X[b,c,l] + beff[o] )
//  - W: Opad x Cpad f16 zero-padded; X: B x Cpad x L f16 zero-padded rows
//  - L multiple of 64 (guaranteed by the pipeline); branch-free inner loop
//  - 128 thr (4 waves): 32(out) x 64(col) tile, 2 v_wmma per wave per K-step
//  - double-buffered LDS: one barrier per K-step, global loads overlap wmma
// ---------------------------------------------------------------------------
template <int RELU, int F32OUT>
__global__ __launch_bounds__(128) void k_gemm_t(
    const f16* __restrict__ W, const f16* __restrict__ X,
    const float* __restrict__ bias,
    f16* __restrict__ Yh, float* __restrict__ Yf,
    int O, int Cpad, int L) {
  __shared__ f16 Wt[2][32][32];   // A tiles, row-major
  __shared__ f16 Xs[2][64][32];   // X tile, column-major: Xs[p][colLocal][k]

  const int b = blockIdx.z;
  const int oBase = blockIdx.y * 32;
  const int colBase = blockIdx.x * 64;
  const int tid = threadIdx.x;
  const int lane = tid & 31;
  const int wave = tid >> 5;
  const int hi = lane >> 4;
  const int lo = lane & 15;

  const f16* Xb = X + (size_t)b * Cpad * L;

  // per-thread tile-fill coordinates
  const int wr = tid >> 2, wc0 = (tid & 3) * 8;   // W: 32x32, b128 per thread
  const int xc = tid & 63, xr = (tid >> 6) * 16;  // X: 64colsx32k, 16 u16/thread

  float4 wreg;
  alignas(16) f16 xreg[16];

  auto loadTile = [&](int kc) {
    wreg = *(const float4*)(W + (size_t)(oBase + wr) * Cpad + kc + wc0);
    const f16* src = Xb + (size_t)(kc + xr) * L + colBase + xc;
#pragma unroll
    for (int e = 0; e < 16; ++e) xreg[e] = src[(size_t)e * L];
  };
  auto storeTile = [&](int p) {
    *(float4*)&Wt[p][wr][wc0]       = wreg;
    *(float4*)&Xs[p][xc][xr]        = *(const float4*)&xreg[0];
    *(float4*)&Xs[p][xc][xr + 8]    = *(const float4*)&xreg[8];
  };
  auto compute = [&](int p, v8f& acc0, v8f& acc1) {
    // contiguous 2x ds_load_b128 per fragment (ISA 7.12.2 wave32 layouts)
    v16h a0 = mk_v16h(*(const float4*)&Wt[p][lo][hi * 8],
                      *(const float4*)&Wt[p][lo][16 + hi * 8]);
    v16h a1 = mk_v16h(*(const float4*)&Wt[p][16 + lo][hi * 8],
                      *(const float4*)&Wt[p][16 + lo][16 + hi * 8]);
    v16h bf = mk_v16h(*(const float4*)&Xs[p][lo + wave * 16][hi * 8],
                      *(const float4*)&Xs[p][lo + wave * 16][16 + hi * 8]);
    acc0 = __builtin_amdgcn_wmma_f32_16x16x32_f16(false, a0, false, bf,
                                                  (short)0, acc0, false, false);
    acc1 = __builtin_amdgcn_wmma_f32_16x16x32_f16(false, a1, false, bf,
                                                  (short)0, acc1, false, false);
  };

  v8f acc0 = {}, acc1 = {};
  loadTile(0);
  storeTile(0);
  __syncthreads();

  int p = 0;
  for (int kc = 32; kc < Cpad; kc += 32) {
    loadTile(kc);            // next-tile global loads issue first
    compute(p, acc0, acc1);  // wmma from current buffer hides load latency
    storeTile(1 - p);
    __syncthreads();
    p ^= 1;
  }
  compute(p, acc0, acc1);

  // ---------------- epilogue (compile-time act/output, uniform O guard) ----
  const int l = colBase + wave * 16 + lo;   // always < L (L % 64 == 0)
  if (oBase + 32 <= O) {                    // full tile: straight-line stores
#pragma unroll
    for (int v = 0; v < 8; ++v) {
      int m = v + (hi << 3);
      float y0 = acc0[v] + bias[oBase + m];
      float y1 = acc1[v] + bias[oBase + 16 + m];
      if (RELU) { y0 = fmaxf(y0, 0.0f); y1 = fmaxf(y1, 0.0f); }
      size_t i0 = ((size_t)b * O + oBase + m) * L + l;
      size_t i1 = ((size_t)b * O + oBase + 16 + m) * L + l;
      if (F32OUT) { Yf[i0] = y0; Yf[i1] = y1; }
      else        { Yh[i0] = (f16)y0; Yh[i1] = (f16)y1; }
    }
  } else {                                  // edge tile (only cls2, O=3)
#pragma unroll
    for (int v = 0; v < 8; ++v) {
      int m = v + (hi << 3);
      int o0 = oBase + m, o1 = oBase + 16 + m;
      if (o0 < O) {
        float y = acc0[v] + bias[o0];
        if (RELU) y = fmaxf(y, 0.0f);
        size_t oi = ((size_t)b * O + o0) * L + l;
        if (F32OUT) Yf[oi] = y; else Yh[oi] = (f16)y;
      }
      if (o1 < O) {
        float y = acc1[v] + bias[o1];
        if (RELU) y = fmaxf(y, 0.0f);
        size_t oi = ((size_t)b * O + o1) * L + l;
        if (F32OUT) Yf[oi] = y; else Yh[oi] = (f16)y;
      }
    }
  }
}

// ---------------------------------------------------------------------------
// Farthest point sampling: one block (256 thr) per batch, dist in global ws.
// ---------------------------------------------------------------------------
__global__ void k_fps(const float* __restrict__ pts, int* __restrict__ outIdx,
                      float* __restrict__ dscr, int N, int M) {
  int b = blockIdx.x;
  const float* px = pts + (size_t)b * 3 * N;
  const float* py = px + N;
  const float* pz = py + N;
  float* dist = dscr + (size_t)b * N;
  __shared__ float sv[256];
  __shared__ int   si[256];
  __shared__ int   lastS;
  int tid = threadIdx.x;
  for (int i = tid; i < N; i += 256) dist[i] = 1e10f;
  if (tid == 0) lastS = 0;
  __syncthreads();
  for (int it = 0; it < M; ++it) {
    int last = lastS;
    if (tid == 0) outIdx[(size_t)b * M + it] = last;
    float lx = px[last], ly = py[last], lz = pz[last];
    float bestV = -1.0f; int bestI = 0;
    for (int i = tid; i < N; i += 256) {
      float dx = px[i] - lx, dy = py[i] - ly, dz = pz[i] - lz;
      float d = dx * dx + dy * dy + dz * dz;
      float dd = dist[i];
      d = d < dd ? d : dd;
      dist[i] = d;
      if (d > bestV) { bestV = d; bestI = i; }
    }
    sv[tid] = bestV; si[tid] = bestI;
    __syncthreads();
    for (int s = 128; s > 0; s >>= 1) {
      if (tid < s) {
        if (sv[tid + s] > sv[tid] ||
            (sv[tid + s] == sv[tid] && si[tid + s] < si[tid])) {
          sv[tid] = sv[tid + s]; si[tid] = si[tid + s];
        }
      }
      __syncthreads();
    }
    if (tid == 0) lastS = si[0];
    __syncthreads();
  }
}

__global__ void k_gather3(const float* __restrict__ pts, const int* __restrict__ idx,
                          float* __restrict__ out, int N, int M, int Bn) {
  int t = blockIdx.x * 256 + threadIdx.x;
  if (t >= Bn * 3 * M) return;
  int m = t % M, c = (t / M) % 3, b = t / (3 * M);
  out[t] = pts[((size_t)b * 3 + c) * N + idx[(size_t)b * M + m]];
}

__global__ void k_ball(const float* __restrict__ pts, const float* __restrict__ newp,
                       int* __restrict__ ind, int N, int M, int K, float r2, int Bn) {
  int t = blockIdx.x * 256 + threadIdx.x;
  if (t >= Bn * M) return;
  int b = t / M, m = t % M;
  const float* px = pts + (size_t)b * 3 * N;
  const float* py = px + N;
  const float* pz = py + N;
  float qx = newp[((size_t)b * 3 + 0) * M + m];
  float qy = newp[((size_t)b * 3 + 1) * M + m];
  float qz = newp[((size_t)b * 3 + 2) * M + m];
  int* o = ind + (size_t)t * K;
  int cnt = 0;
  for (int n = 0; n < N && cnt < K; ++n) {
    float dx = px[n] - qx, dy = py[n] - qy, dz = pz[n] - qz;
    if (dx * dx + dy * dy + dz * dz < r2) o[cnt++] = n;
  }
  if (cnt == 0) { for (int kk = 0; kk < K; ++kk) o[kk] = N - 1; }
  else { int f = o[0]; for (int kk = cnt; kk < K; ++kk) o[kk] = f; }
}

// kNN (K<=64), insertion sort, stable on ties (matches top_k ordering).
__global__ void k_knn(const float* __restrict__ pts, const float* __restrict__ qry,
                      int* __restrict__ ind, float* __restrict__ odist,
                      int N, int M, int K, int Bn) {
  int t = blockIdx.x * 256 + threadIdx.x;
  if (t >= Bn * M) return;
  int b = t / M, m = t % M;
  const float* px = pts + (size_t)b * 3 * N;
  const float* py = px + N;
  const float* pz = py + N;
  float qx = qry[((size_t)b * 3 + 0) * M + m];
  float qy = qry[((size_t)b * 3 + 1) * M + m];
  float qz = qry[((size_t)b * 3 + 2) * M + m];
  float bd[64]; int bi[64];
  int cnt = 0;
  for (int n = 0; n < N; ++n) {
    float dx = px[n] - qx, dy = py[n] - qy, dz = pz[n] - qz;
    float d = dx * dx + dy * dy + dz * dz;
    if (cnt < K) {
      int j = cnt++;
      while (j > 0 && bd[j - 1] > d) { bd[j] = bd[j - 1]; bi[j] = bi[j - 1]; --j; }
      bd[j] = d; bi[j] = n;
    } else if (d < bd[K - 1]) {
      int j = K - 1;
      while (j > 0 && bd[j - 1] > d) { bd[j] = bd[j - 1]; bi[j] = bi[j - 1]; --j; }
      bd[j] = d; bi[j] = n;
    }
  }
  int* o = ind + (size_t)t * K;
  for (int kk = 0; kk < K; ++kk) o[kk] = bi[kk];
  if (odist) {
    float* od = odist + (size_t)t * K;
    for (int kk = 0; kk < K; ++kk) od[kk] = bd[kk];
  }
}

// Build grouped input (B, Cpad, M, K): coord diffs, gathered featG, bcast featB,
// zero fill for channels in [3+C1+C2, Cpad).
__global__ void k_group(const float* __restrict__ pts, const float* __restrict__ newp,
                        const f16* __restrict__ featG, const f16* __restrict__ featB,
                        const int* __restrict__ ind, f16* __restrict__ out,
                        int N, int M, int K, int C1, int C2, int Cpad, int Bn) {
  int Ctot = 3 + C1 + C2;
  long long t = (long long)blockIdx.x * 256 + threadIdx.x;
  long long total = (long long)Bn * Cpad * M * K;
  if (t >= total) return;
  int k = (int)(t % K);
  int m = (int)((t / K) % M);
  int c = (int)((t / ((long long)K * M)) % Cpad);
  int b = (int)(t / ((long long)K * M * Cpad));
  f16 v;
  if (c >= Ctot) {
    v = (f16)0.0f;
  } else {
    int id = ind[((size_t)b * M + m) * K + k];
    if (c < 3)
      v = (f16)(pts[((size_t)b * 3 + c) * N + id] - newp[((size_t)b * 3 + c) * M + m]);
    else if (c < 3 + C1)
      v = featG[((size_t)b * C1 + (c - 3)) * N + id];
    else
      v = featB[((size_t)b * C2 + (c - 3 - C1)) * M + m];
  }
  out[t] = v;
}

// max over K; input (B, Cpadin, M, K), output (B, C, M) dense
__global__ void k_maxpool(const f16* __restrict__ in, f16* __restrict__ out,
                          int C, int Cpadin, int M, int K, int Bn) {
  int t = blockIdx.x * 256 + threadIdx.x;
  if (t >= Bn * C * M) return;
  int m = t % M, c = (t / M) % C, b = t / (M * C);
  const f16* p = in + (((size_t)b * Cpadin + c) * M + m) * K;
  float mx = -3.0e38f;
  for (int k = 0; k < K; ++k) { float v = (float)p[k]; if (v > mx) mx = v; }
  out[((size_t)b * C + c) * M + m] = (f16)mx;
}

// concat channels with zero pad: out (B, Cpad, L)
__global__ void k_concat(const f16* __restrict__ A, const f16* __restrict__ Bp,
                         f16* __restrict__ out, int C1, int C2, int Cpad,
                         int L, int Bn) {
  long long t = (long long)blockIdx.x * 256 + threadIdx.x;
  long long total = (long long)Bn * Cpad * L;
  if (t >= total) return;
  int l = (int)(t % L);
  int c = (int)((t / L) % Cpad);
  int b = (int)(t / ((long long)L * Cpad));
  f16 v;
  if (c < C1)           v = A[((size_t)b * C1 + c) * L + l];
  else if (c < C1 + C2) v = Bp[((size_t)b * C2 + (c - C1)) * L + l];
  else                  v = (f16)0.0f;
  out[t] = v;
}

// 3-NN inverse-distance interpolation
__global__ void k_interp(const f16* __restrict__ f1, const int* __restrict__ ind,
                         const float* __restrict__ dist, f16* __restrict__ out,
                         int C, int Nsrc, int Nq, int Bn) {
  long long t = (long long)blockIdx.x * 256 + threadIdx.x;
  long long total = (long long)Bn * C * Nq;
  if (t >= total) return;
  int q = (int)(t % Nq);
  int c = (int)((t / Nq) % C);
  int b = (int)(t / ((long long)Nq * C));
  const int*   ii = ind  + ((size_t)b * Nq + q) * 3;
  const float* dd = dist + ((size_t)b * Nq + q) * 3;
  float w0 = 1.0f / fmaxf(dd[0], 1e-10f);
  float w1 = 1.0f / fmaxf(dd[1], 1e-10f);
  float w2 = 1.0f / fmaxf(dd[2], 1e-10f);
  float s = w0 + w1 + w2;
  const f16* fb = f1 + ((size_t)b * C + c) * Nsrc;
  float v = ((float)fb[ii[0]] * w0 + (float)fb[ii[1]] * w1 + (float)fb[ii[2]] * w2) / s;
  out[t] = (f16)v;
}

// ---------------------------------------------------------------------------
// Host orchestration
// ---------------------------------------------------------------------------
struct LayerSpec { int pbase; int cin; int cout; };
static const LayerSpec LS[29] = {
  {4,6,32},{10,32,32},{16,32,64},          // 0-2  sc1
  {22,67,64},{28,64,64},{34,64,128},       // 3-5  sc2
  {40,259,128},{46,128,128},{52,128,128},  // 6-8  fe
  {58,131,128},{64,128,128},{70,128,256},  // 9-11 sc3
  {76,259,256},{82,256,256},{88,256,512},  //12-14 sc4
  {94,771,256},{100,256,256},              //15-16 up1_c2
  {106,259,128},{112,128,128},{118,128,256},//17-19 up2_c1
  {124,512,256},                           //20    up2_c2
  {130,259,128},{136,128,128},{142,128,256},//21-23 up3_c1
  {148,320,256},                           //24    up3_c2
  {154,259,256},{160,256,256},             //25-26 fp
  {166,256,128},                           //27    cls1
  {172,128,3},                             //28    cls2 (no BN)
};

extern "C" void kernel_launch(void* const* d_in, const int* in_sizes, int n_in,
                              void* d_out, int out_size, void* d_ws, size_t ws_size,
                              hipStream_t stream) {
  if (n_in < 174) return;
  const int N0 = 4096;
  const int Bb = in_sizes[0] / (3 * N0);

  auto IN = [&](int i) -> const float* { return (const float*)d_in[i]; };
  char* ws = (char*)d_ws;
  size_t off = 0;
  auto alloc = [&](size_t bytes) -> size_t {
    off = (off + 255) & ~(size_t)255;
    size_t r = off; off += bytes; return r;
  };
  auto F32 = [&](size_t n) -> float* { return (float*)(ws + alloc(n * 4)); };
  auto H16 = [&](size_t n) -> f16*   { return (f16*)(ws + alloc(n * 2)); };
  auto I32 = [&](size_t n) -> int*   { return (int*)(ws + alloc(n * 4)); };

  // folded, padded params
  size_t wOff[29], bOff[29];
  for (int i = 0; i < 29; ++i) {
    int Opad = cp32(LS[i].cout), Cpad = cp32(LS[i].cin);
    wOff[i] = alloc((size_t)Opad * Cpad * sizeof(f16));
    bOff[i] = alloc((size_t)LS[i].cout * sizeof(float));
  }
  float* dscr    = F32((size_t)Bb * N0);
  int*   fpsIdx  = I32((size_t)Bb * 1024);
  int*   indBuf  = I32((size_t)Bb * 16384);
  float* distBuf = F32((size_t)Bb * N0 * 3);
  float* p11a = F32((size_t)Bb * 3 * 1024);
  float* p21  = F32((size_t)Bb * 3 * 1024);
  float* p12a = F32((size_t)Bb * 3 * 256);
  float* p22  = F32((size_t)Bb * 3 * 256);
  float* p13  = F32((size_t)Bb * 3 * 64);
  float* p14  = F32((size_t)Bb * 3 * 16);
  f16* feat1h = H16((size_t)Bb * 3 * N0);
  f16* feat2h = H16((size_t)Bb * 3 * N0);
  f16* f11a = H16((size_t)Bb * 64 * 1024);
  f16* f21  = H16((size_t)Bb * 64 * 1024);
  f16* f12a = H16((size_t)Bb * 128 * 256);
  f16* f22  = H16((size_t)Bb * 128 * 256);
  f16* emb  = H16((size_t)Bb * 128 * 256);
  f16* f13  = H16((size_t)Bb * 256 * 64);
  f16* f14  = H16((size_t)Bb * 512 * 16);
  f16* nf13 = H16((size_t)Bb * 256 * 64);
  f16* nf12 = H16((size_t)Bb * 256 * 256);
  f16* nf11 = H16((size_t)Bb * 256 * 1024);
  f16* cat256 = H16((size_t)Bb * 256 * 256);
  f16* poolB  = H16((size_t)Bb * 256 * 4096);
  f16* xbuf = H16((size_t)Bb * 288 * 16384);   // max Cpad*L across stages
  f16* t0   = H16((size_t)Bb * 128 * 16384);
  f16* t1   = H16((size_t)Bb * 128 * 16384);
  (void)ws_size;

  // fold BN into padded weights
  for (int i = 0; i < 29; ++i) {
    int Opad = cp32(LS[i].cout), Cpad = cp32(LS[i].cin);
    int tot = Opad * Cpad;
    f16* We = (f16*)(ws + wOff[i]);
    float* be = (float*)(ws + bOff[i]);
    int pb = LS[i].pbase;
    if (i == 28)
      k_fold_plain<<<ceilDiv(tot, 256), 256, 0, stream>>>(IN(pb), IN(pb + 1), We, be,
                                                          LS[i].cout, LS[i].cin,
                                                          Cpad, tot);
    else
      k_fold<<<ceilDiv(tot, 256), 256, 0, stream>>>(IN(pb), IN(pb + 1), IN(pb + 2),
                                                    IN(pb + 3), IN(pb + 4), IN(pb + 5),
                                                    We, be, LS[i].cout, LS[i].cin,
                                                    Cpad, tot);
  }
  // f32 -> f16 input features
  k_f2h<<<ceilDiv(Bb * 3 * N0, 256), 256, 0, stream>>>(IN(2), feat1h, Bb * 3 * N0);
  k_f2h<<<ceilDiv(Bb * 3 * N0, 256), 256, 0, stream>>>(IN(3), feat2h, Bb * 3 * N0);

  auto runLayer = [&](int li, const f16* Xin, int L, f16* Yh, float* Yf, int relu) {
    dim3 g(L / 64, ceilDiv(LS[li].cout, 32), Bb);
    const f16* Wp = (const f16*)(ws + wOff[li]);
    const float* bp = (const float*)(ws + bOff[li]);
    int O = LS[li].cout, Cpad = cp32(LS[li].cin);
    if (Yf) {
      if (relu) k_gemm_t<1, 1><<<g, 128, 0, stream>>>(Wp, Xin, bp, Yh, Yf, O, Cpad, L);
      else      k_gemm_t<0, 1><<<g, 128, 0, stream>>>(Wp, Xin, bp, Yh, Yf, O, Cpad, L);
    } else {
      if (relu) k_gemm_t<1, 0><<<g, 128, 0, stream>>>(Wp, Xin, bp, Yh, Yf, O, Cpad, L);
      else      k_gemm_t<0, 0><<<g, 128, 0, stream>>>(Wp, Xin, bp, Yh, Yf, O, Cpad, L);
    }
  };
  auto runChainPP = [&](const int* ls, int n, const f16* Xin, int L) -> const f16* {
    const f16* cur = Xin;
    for (int i = 0; i < n; ++i) {
      f16* o = (cur == t0) ? t1 : t0;
      runLayer(ls[i], cur, L, o, nullptr, 1);
      cur = o;
    }
    return cur;
  };

  auto setConv = [&](const float* ptsIn, int Nin, const f16* featIn, int Cin,
                     int Mout, float r2, int K, const int* layers, int nL,
                     float* newp, f16* featOut, int Cout) {
    k_fps<<<Bb, 256, 0, stream>>>(ptsIn, fpsIdx, dscr, Nin, Mout);
    k_gather3<<<ceilDiv(Bb * 3 * Mout, 256), 256, 0, stream>>>(ptsIn, fpsIdx, newp,
                                                               Nin, Mout, Bb);
    k_ball<<<ceilDiv(Bb * Mout, 256), 256, 0, stream>>>(ptsIn, newp, indBuf,
                                                        Nin, Mout, K, r2, Bb);
    int Cpad = cp32(3 + Cin);
    long long tot = (long long)Bb * Cpad * Mout * K;
    k_group<<<(int)((tot + 255) / 256), 256, 0, stream>>>(ptsIn, newp, featIn, nullptr,
                                                          indBuf, xbuf, Nin, Mout, K,
                                                          Cin, 0, Cpad, Bb);
    const f16* last = runChainPP(layers, nL, xbuf, Mout * K);
    k_maxpool<<<ceilDiv(Bb * Cout * Mout, 256), 256, 0, stream>>>(last, featOut,
                                                                  Cout, Cout, Mout,
                                                                  K, Bb);
  };

  auto setUpconv = [&](const float* p1, int N1, const float* p2, int M2,
                       const f16* f1, int C1, const f16* f2b, int C2f, int K,
                       const int* c1L, int nC1, const int* c2L, int nC2, f16* dst) {
    k_knn<<<ceilDiv(Bb * M2, 256), 256, 0, stream>>>(p1, p2, indBuf, nullptr,
                                                     N1, M2, K, Bb);
    int CpadG = cp32(3 + C1);
    long long tot = (long long)Bb * CpadG * M2 * K;
    k_group<<<(int)((tot + 255) / 256), 256, 0, stream>>>(p1, p2, f1, nullptr, indBuf,
                                                          xbuf, N1, M2, K, C1, 0,
                                                          CpadG, Bb);
    const f16* pre; int Cpre, CpreIn;
    if (nC1 > 0) {
      pre = runChainPP(c1L, nC1, xbuf, M2 * K);
      Cpre = LS[c1L[nC1 - 1]].cout; CpreIn = Cpre;
    } else { pre = xbuf; Cpre = 3 + C1; CpreIn = CpadG; }
    k_maxpool<<<ceilDiv(Bb * Cpre * M2, 256), 256, 0, stream>>>(pre, poolB,
                                                                Cpre, CpreIn, M2,
                                                                K, Bb);
    int CpadC = cp32(Cpre + C2f);
    long long ct = (long long)Bb * CpadC * M2;
    k_concat<<<(int)((ct + 255) / 256), 256, 0, stream>>>(poolB, f2b, xbuf,
                                                          Cpre, C2f, CpadC, M2, Bb);
    const f16* cur = xbuf;
    for (int i = 0; i < nC2; ++i) {
      f16* o = (i == nC2 - 1) ? dst : ((cur == t0) ? t1 : t0);
      runLayer(c2L[i], cur, M2, o, nullptr, 1);
      cur = o;
    }
  };

  static const int sc1L[3] = {0, 1, 2}, sc2L[3] = {3, 4, 5}, feL[3] = {6, 7, 8};
  static const int sc3L[3] = {9, 10, 11}, sc4L[3] = {12, 13, 14};
  static const int up1c2L[2] = {15, 16};
  static const int up2c1L[3] = {17, 18, 19}, up2c2L[1] = {20};
  static const int up3c1L[3] = {21, 22, 23}, up3c2L[1] = {24};

  // ---- branch 1 ----
  setConv(IN(0), N0, feat1h, 3, 1024, 0.25f, 16, sc1L, 3, p11a, f11a, 64);
  setConv(p11a, 1024, f11a, 64, 256, 1.0f, 16, sc2L, 3, p12a, f12a, 128);
  // ---- branch 2 ----
  setConv(IN(1), N0, feat2h, 3, 1024, 0.25f, 16, sc1L, 3, p21, f21, 64);
  setConv(p21, 1024, f21, 64, 256, 1.0f, 16, sc2L, 3, p22, f22, 128);

  // ---- flow embedding (k=64 kNN of p12 queries in p22) ----
  k_knn<<<ceilDiv(Bb * 256, 256), 256, 0, stream>>>(p22, p12a, indBuf, nullptr,
                                                    256, 256, 64, Bb);
  {
    int Cpad = cp32(259);  // 288
    long long tot = (long long)Bb * Cpad * 256 * 64;
    k_group<<<(int)((tot + 255) / 256), 256, 0, stream>>>(p22, p12a, f22, f12a, indBuf,
                                                          xbuf, 256, 256, 64, 128, 128,
                                                          Cpad, Bb);
    const f16* last = runChainPP(feL, 3, xbuf, 256 * 64);
    k_maxpool<<<ceilDiv(Bb * 128 * 256, 256), 256, 0, stream>>>(last, emb,
                                                                128, 128, 256, 64, Bb);
  }

  // ---- sc3 / sc4 ----
  setConv(p12a, 256, emb, 128, 64, 4.0f, 8, sc3L, 3, p13, f13, 256);
  setConv(p13, 64, f13, 256, 16, 16.0f, 8, sc4L, 3, p14, f14, 512);

  // ---- up-convolutions ----
  setUpconv(p14, 16, p13, 64, f14, 512, f13, 256, 8, nullptr, 0, up1c2L, 2, nf13);
  {
    int Cpad = 256;
    long long ct = (long long)Bb * Cpad * 256;
    k_concat<<<(int)((ct + 255) / 256), 256, 0, stream>>>(f12a, emb, cat256,
                                                          128, 128, Cpad, 256, Bb);
  }
  setUpconv(p13, 64, p12a, 256, nf13, 256, cat256, 256, 8, up2c1L, 3, up2c2L, 1, nf12);
  setUpconv(p12a, 256, p11a, 1024, nf12, 256, f11a, 64, 8, up3c1L, 3, up3c2L, 1, nf11);

  // ---- feature propagation + classifier ----
  k_knn<<<ceilDiv(Bb * N0, 256), 256, 0, stream>>>(p11a, IN(0), indBuf, distBuf,
                                                   1024, N0, 3, Bb);
  {
    long long tot = (long long)Bb * 256 * N0;
    k_interp<<<(int)((tot + 255) / 256), 256, 0, stream>>>(nf11, indBuf, distBuf, poolB,
                                                           256, 1024, N0, Bb);
    int Cpad = cp32(259);  // 288
    long long ct = (long long)Bb * Cpad * N0;
    k_concat<<<(int)((ct + 255) / 256), 256, 0, stream>>>(poolB, feat1h, xbuf,
                                                          256, 3, Cpad, N0, Bb);
  }
  runLayer(25, xbuf, N0, t0, nullptr, 1);   // fp 259->256
  runLayer(26, t0, N0, t1, nullptr, 1);     // fp 256->256
  runLayer(27, t1, N0, t0, nullptr, 1);     // cls1 256->128
  runLayer(28, t0, N0, nullptr, (float*)d_out, 0);  // cls2 128->3, f32 out
  (void)out_size;
}
